// TransformerModel_58463094833742
// MI455X (gfx1250) — matrix operand order repaired and checked by reading, not compiled, not run on hardware
//
#include <hip/hip_runtime.h>
#include <math.h>

// Model dims (fixed by the reference)
#define SEQ    2048
#define DMODEL 512
#define NHEAD  8
#define DHEAD  64
#define NLAYER 6
#define DFF    2048
#define NVOCAB 50257
#define VPAD   50304   // NVOCAB rounded up to 64

typedef __attribute__((ext_vector_type(16))) _Float16 v16h;
typedef __attribute__((ext_vector_type(8)))  float    v8f;
typedef int v4i __attribute__((vector_size(16)));   // b128 payload type

// Async global->LDS DMA path (gfx1250), with compile-safe fallback
#if defined(__gfx1250__) && __has_builtin(__builtin_amdgcn_global_load_async_to_lds_b128) && __has_builtin(__builtin_amdgcn_s_wait_asynccnt)
#define USE_ASYNC_LDS 1
#else
#define USE_ASYNC_LDS 0
#endif

// B-operand modes
#define BM_F32N 0   // B fp32 [K,N] row-major (P @ V)
#define BM_F32T 1   // B fp32 [N,K] row-major (Q @ K^T)
#define BM_F16T 2   // B f16  [N,K] pre-transposed weights (async-to-LDS)

// ---------------------------------------------------------------------------
// Epilogue store helper (all flags compile-time)
// ---------------------------------------------------------------------------
template<int ACT, int RES, int GUARD>
__device__ __forceinline__ void store_one(
    float* __restrict__ C, const float* __restrict__ Res,
    float v, float bv, int m, int n, int N, int ldc)
{
    if (GUARD && n >= N) return;
    float x = v + bv;
    if (RES) x += Res[(size_t)m * ldc + n];
    if (ACT) x = 0.5f * x * (1.0f + erff(x * 0.70710678118654752f));
    C[(size_t)m * ldc + n] = x;
}

// ---------------------------------------------------------------------------
// WMMA GEMM:  C[M,N] = act( A[M,K] @ op(B) + bias + residual )
//   - A fp32, converted to f16 while staging into LDS
//   - B per BMODE; BM_F16T rows are DMA'd to LDS via
//     global_load_async_to_lds_b128 (ASYNCcnt) when available
//   - fp32 accumulate via v_wmma_f32_16x16x32_f16
//   - block tile 128x64, K-step 32; 8 waves as 4(M) x 2(N); each wave owns a
//     32x32 tile = 4 accumulators, 4 WMMAs per K-step
//   - GUARD: epilogue N-bounds only (LM head); B loads are padded in ws
// ---------------------------------------------------------------------------
template<int BMODE, int ACT, int BIAS, int RES, int GUARD>
__global__ __launch_bounds__(256) void gemm_wmma(
    const float* __restrict__ A,
    const float* __restrict__ Bf, const _Float16* __restrict__ Bh,
    const float* __restrict__ bias, const float* __restrict__ Res,
    float* __restrict__ C,
    int M, int N, int K, int lda, int ldb, int ldc)
{
    __shared__ _Float16 As[128][40];  // [m][k]; 80B row stride
    __shared__ _Float16 Bs[64][40];   // [n][k]; 80B row stride (16B-aligned segs)

    const int tid  = threadIdx.x;
    const int m0   = blockIdx.y * 128;
    const int n0   = blockIdx.x * 64;
    const int w    = tid >> 5;        // wave 0..7 (wave32)
    const int lane = tid & 31;
    const int wy   = w >> 1;          // 0..3 -> 32-row M sub-block
    const int wx   = w & 1;           // 0..1 -> 32-col N sub-block
    const int laneRow = lane & 15;
    const int khi     = lane >> 4;    // lane half selects K interleave

    // A staging: 16 consecutive halves per thread (coalesced)
    const int aRow = tid >> 1;            // 0..127
    const int aK   = (tid & 1) * 16;      // 0 / 16
    // B staging (BM_F16T / BM_F32T): n-major, contiguous k
    const int bnRow = tid >> 2;           // 0..63
    const int bnK   = (tid & 3) * 8;      // 0,8,16,24
    // B staging (BM_F32N): k-major, contiguous n (coalesced), scatter to [n][k]
    const int bkRow = tid >> 3;           // 0..31
    const int bkN   = (tid & 7) * 8;

    v8f acc00 = {}, acc01 = {}, acc10 = {}, acc11 = {};

    for (int k0 = 0; k0 < K; k0 += 32) {
        // ---- stage A tile (128x32), fp32 -> f16 ----
        const float* Ap = A + (size_t)(m0 + aRow) * lda + (k0 + aK);
        #pragma unroll
        for (int j = 0; j < 16; ++j)
            As[aRow][aK + j] = (_Float16)Ap[j];
        if (k0 + 32 < K)
            __builtin_prefetch(Ap + 32, 0, 1);

        // ---- stage B tile into [n][k] ----
        if (BMODE == BM_F16T) {
            const _Float16* Bp = Bh + (size_t)(n0 + bnRow) * ldb + (k0 + bnK);
#if USE_ASYNC_LDS
            __builtin_amdgcn_global_load_async_to_lds_b128(
                (__attribute__((address_space(1))) v4i*)Bp,
                (__attribute__((address_space(3))) v4i*)&Bs[bnRow][bnK],
                0, 0);
#else
            #pragma unroll
            for (int j = 0; j < 8; ++j)
                Bs[bnRow][bnK + j] = Bp[j];
#endif
        } else if (BMODE == BM_F32T) {
            const float* Bp = Bf + (size_t)(n0 + bnRow) * ldb + (k0 + bnK);
            #pragma unroll
            for (int j = 0; j < 8; ++j)
                Bs[bnRow][bnK + j] = (_Float16)Bp[j];
            if (k0 + 32 < K)
                __builtin_prefetch(Bp + 32, 0, 1);
        } else {
            const float* Bp = Bf + (size_t)(k0 + bkRow) * ldb + (n0 + bkN);
            #pragma unroll
            for (int j = 0; j < 8; ++j)
                Bs[bkN + j][bkRow] = (_Float16)Bp[j];
            if (k0 + 32 < K)
                __builtin_prefetch(Bp + (size_t)32 * ldb, 0, 1);
        }

#if USE_ASYNC_LDS
        if (BMODE == BM_F16T)
            __builtin_amdgcn_s_wait_asynccnt(0);
#endif
        __syncthreads();

        // ---- gather fragments per ISA 7.12.2 layouts (aligned dwords) ----
        union { v16h h; unsigned int u[8]; } fa0, fa1, fb0, fb1;
        #pragma unroll
        for (int v = 0; v < 8; ++v) {
            const int ks = (v < 4) ? (2 * v + 8 * khi)
                                   : (8 + 2 * v + 8 * khi);   // 16 + 2*(v-4)
            fa0.u[v] = *(const unsigned int*)&As[wy * 32 +      laneRow][ks];
            fa1.u[v] = *(const unsigned int*)&As[wy * 32 + 16 + laneRow][ks];
            fb0.u[v] = *(const unsigned int*)&Bs[wx * 32 +      laneRow][ks];
            fb1.u[v] = *(const unsigned int*)&Bs[wx * 32 + 16 + laneRow][ks];
        }

        acc00 = __builtin_amdgcn_wmma_f32_16x16x32_f16(
                    false, fa0.h, false, fb0.h, (short)0, acc00, false, false);
        acc01 = __builtin_amdgcn_wmma_f32_16x16x32_f16(
                    false, fa0.h, false, fb1.h, (short)0, acc01, false, false);
        acc10 = __builtin_amdgcn_wmma_f32_16x16x32_f16(
                    false, fa1.h, false, fb0.h, (short)0, acc10, false, false);
        acc11 = __builtin_amdgcn_wmma_f32_16x16x32_f16(
                    false, fa1.h, false, fb1.h, (short)0, acc11, false, false);
        __syncthreads();
    }

    // ---- epilogue ----
    const int nA = n0 + wx * 32 + laneRow;
    const int nB = nA + 16;
    float bvA = 0.0f, bvB = 0.0f;
    if (BIAS) {
        if (!GUARD || nA < N) bvA = bias[nA];
        if (!GUARD || nB < N) bvB = bias[nB];
    }
    #pragma unroll
    for (int r = 0; r < 8; ++r) {
        const int mA = m0 + wy * 32 + khi * 8 + r;   // VGPR r -> M = r + 8*khi
        const int mB = mA + 16;
        store_one<ACT, RES, GUARD>(C, Res, acc00[r], bvA, mA, nA, N, ldc);
        store_one<ACT, RES, GUARD>(C, Res, acc01[r], bvB, mA, nB, N, ldc);
        store_one<ACT, RES, GUARD>(C, Res, acc10[r], bvA, mB, nA, N, ldc);
        store_one<ACT, RES, GUARD>(C, Res, acc11[r], bvB, mB, nB, N, ldc);
    }
}

// ---------------------------------------------------------------------------
// Tiled transpose + fp32->f16: W[K,N] -> Wt[Npad][K]; pad rows zero-filled
// grid (Npad/64, K/64), 256 threads; K multiple of 64 in all uses
// ---------------------------------------------------------------------------
__global__ __launch_bounds__(256) void transpose_to_f16(
    const float* __restrict__ W, _Float16* __restrict__ Wt,
    int K, int N, int Npad)
{
    __shared__ _Float16 t[64][65];
    const int tid = threadIdx.x;
    const int nin = blockIdx.x * 64;
    const int kin = blockIdx.y * 64;

    const int row  = tid >> 2;          // 0..63
    const int col4 = (tid & 3) * 16;    // 16 consecutive elements

    // read W rows (coalesced), store transposed into LDS tile [n][k]
    const float* Wp = W + (size_t)(kin + row) * N + (nin + col4);
    #pragma unroll
    for (int j = 0; j < 16; ++j) {
        const int n = nin + col4 + j;
        t[col4 + j][row] = (_Float16)((n < N) ? Wp[j] : 0.0f);
    }
    __syncthreads();

    // write Wt rows (coalesced)
    _Float16* Op = Wt + (size_t)(nin + row) * K + (kin + col4);
    #pragma unroll
    for (int j = 0; j < 16; ++j)
        Op[j] = t[row][col4 + j];
}

// ---------------------------------------------------------------------------
// Token embedding * sqrt(D) + sinusoidal positional encoding
// ---------------------------------------------------------------------------
__global__ __launch_bounds__(256) void embed_pe(
    const int* __restrict__ x, const float* __restrict__ emb,
    float* __restrict__ h)
{
    const int idx = blockIdx.x * 256 + threadIdx.x;   // over S*D
    const int s = idx / DMODEL;
    const int d = idx - s * DMODEL;
    const int tok = x[s];
    const float freq = __expf(-(float)(d & ~1) * (9.210340371976184f / (float)DMODEL));
    const float ang  = (float)s * freq;
    const float pe   = (d & 1) ? __cosf(ang) : __sinf(ang);
    h[idx] = emb[(size_t)tok * DMODEL + d] * 22.627416997969522f /* sqrt(512) */ + pe;
}

// ---------------------------------------------------------------------------
// Causal, scaled, row-wise softmax (in-place)
// ---------------------------------------------------------------------------
__global__ __launch_bounds__(256) void softmax_causal(
    float* __restrict__ scores, float scale)
{
    const int q = blockIdx.x;
    float* row = scores + (size_t)q * SEQ;
    const int nvalid = q + 1;
    __shared__ float red[256];
    const int tid = threadIdx.x;

    float mx = -3.4e38f;
    for (int i = tid; i < nvalid; i += 256) mx = fmaxf(mx, row[i] * scale);
    red[tid] = mx; __syncthreads();
    for (int s = 128; s > 0; s >>= 1) {
        if (tid < s) red[tid] = fmaxf(red[tid], red[tid + s]);
        __syncthreads();
    }
    mx = red[0]; __syncthreads();

    float sum = 0.0f;
    for (int i = tid; i < nvalid; i += 256) {
        const float e = expf(row[i] * scale - mx);
        row[i] = e;
        sum += e;
    }
    red[tid] = sum; __syncthreads();
    for (int s = 128; s > 0; s >>= 1) {
        if (tid < s) red[tid] += red[tid + s];
        __syncthreads();
    }
    const float inv = 1.0f / red[0];
    for (int i = tid; i < nvalid; i += 256) row[i] *= inv;
    for (int i = nvalid + tid; i < SEQ; i += 256) row[i] = 0.0f;
}

// ---------------------------------------------------------------------------
// LayerNorm over last dim (D=512)
// ---------------------------------------------------------------------------
__global__ __launch_bounds__(256) void layernorm_row(
    const float* __restrict__ in, const float* __restrict__ g,
    const float* __restrict__ b, float* __restrict__ out)
{
    const int s = blockIdx.x;
    const float* row = in + (size_t)s * DMODEL;
    __shared__ float red[256];
    const int tid = threadIdx.x;

    float sum = 0.0f;
    for (int i = tid; i < DMODEL; i += 256) sum += row[i];
    red[tid] = sum; __syncthreads();
    for (int st = 128; st > 0; st >>= 1) {
        if (tid < st) red[tid] += red[tid + st];
        __syncthreads();
    }
    const float mean = red[0] * (1.0f / DMODEL);
    __syncthreads();

    float vs = 0.0f;
    for (int i = tid; i < DMODEL; i += 256) {
        const float d = row[i] - mean;
        vs += d * d;
    }
    red[tid] = vs; __syncthreads();
    for (int st = 128; st > 0; st >>= 1) {
        if (tid < st) red[tid] += red[tid + st];
        __syncthreads();
    }
    const float inv = rsqrtf(red[0] * (1.0f / DMODEL) + 1e-5f);
    for (int i = tid; i < DMODEL; i += 256)
        out[(size_t)s * DMODEL + i] = (row[i] - mean) * inv * g[i] + b[i];
}

// ---------------------------------------------------------------------------
// Host orchestration
// ---------------------------------------------------------------------------
#define GEMM(BM, AC, BI, RE, GU, Ap, Bfp, Bhp, biasp, Resp, Cp, M_, N_, K_, lda_, ldb_, ldc_) \
    gemm_wmma<BM, AC, BI, RE, GU>                                                             \
        <<<dim3((unsigned)(((N_) + 63) / 64), (unsigned)((M_) / 128)), 256, 0, stream>>>      \
        (Ap, Bfp, Bhp, biasp, Resp, Cp, M_, N_, K_, lda_, ldb_, ldc_)

extern "C" void kernel_launch(void* const* d_in, const int* in_sizes, int n_in,
                              void* d_out, int out_size, void* d_ws, size_t ws_size,
                              hipStream_t stream)
{
    (void)in_sizes; (void)n_in; (void)out_size; (void)ws_size;
    const int S = SEQ, D = DMODEL, F = DFF, V = NVOCAB, L = NLAYER;

    const int*   x     = (const int*)  d_in[0];
    /* d_in[1] mask: causal tril, implemented analytically */
    const float* emb   = (const float*)d_in[2];
    const float* wq    = (const float*)d_in[3];
    const float* bq    = (const float*)d_in[4];
    const float* wk    = (const float*)d_in[5];
    const float* bk    = (const float*)d_in[6];
    const float* wv    = (const float*)d_in[7];
    const float* bv    = (const float*)d_in[8];
    const float* wo    = (const float*)d_in[9];
    const float* bo    = (const float*)d_in[10];
    const float* w1    = (const float*)d_in[11];
    const float* b1    = (const float*)d_in[12];
    const float* w2    = (const float*)d_in[13];
    const float* b2    = (const float*)d_in[14];
    const float* ln1g  = (const float*)d_in[15];
    const float* ln1b  = (const float*)d_in[16];
    const float* ln2g  = (const float*)d_in[17];
    const float* ln2b  = (const float*)d_in[18];
    const float* lnfg  = (const float*)d_in[19];
    const float* lnfb  = (const float*)d_in[20];
    const float* headw = (const float*)d_in[21];
    const float* headb = (const float*)d_in[22];
    float*       out   = (float*)d_out;

    // ---- workspace layout: ~56 MB fp32 + ~92 MB f16 weights = ~148 MB ----
    float* ws  = (float*)d_ws;
    float* h   = ws;                  // [S,D] running hidden state
    float* tmp = h   + (size_t)S * D; // [S,D] pre-LN residual sums
    float* q   = tmp + (size_t)S * D; // [S,D]
    float* k   = q   + (size_t)S * D; // [S,D]
    float* v   = k   + (size_t)S * D; // [S,D]
    float* ao  = v   + (size_t)S * D; // [S,D] attention output (pre O-proj)
    float* ff  = ao  + (size_t)S * D; // [S,F]
    float* sc  = ff  + (size_t)S * F; // [S,S] per-head scores (reused)

    _Float16* wqT   = (_Float16*)(sc + (size_t)S * S);     // [L][D][D]
    _Float16* wkT   = wqT + (size_t)L * D * D;             // [L][D][D]
    _Float16* wvT   = wkT + (size_t)L * D * D;             // [L][D][D]
    _Float16* woT   = wvT + (size_t)L * D * D;             // [L][D][D]
    _Float16* w1T   = woT + (size_t)L * D * D;             // [L][F][D]
    _Float16* w2T   = w1T + (size_t)L * F * D;             // [L][D][F]
    _Float16* headT = w2T + (size_t)L * D * F;             // [VPAD][D]

    const float scale = 0.125f; // 1/sqrt(DK)

    // ---- one pass: transpose + fp32->f16 all weights ----
    for (int l = 0; l < L; ++l) {
        dim3 gdd(D / 64, D / 64);
        transpose_to_f16<<<gdd, 256, 0, stream>>>(wq + (size_t)l * D * D, wqT + (size_t)l * D * D, D, D, D);
        transpose_to_f16<<<gdd, 256, 0, stream>>>(wk + (size_t)l * D * D, wkT + (size_t)l * D * D, D, D, D);
        transpose_to_f16<<<gdd, 256, 0, stream>>>(wv + (size_t)l * D * D, wvT + (size_t)l * D * D, D, D, D);
        transpose_to_f16<<<gdd, 256, 0, stream>>>(wo + (size_t)l * D * D, woT + (size_t)l * D * D, D, D, D);
        transpose_to_f16<<<dim3(F / 64, D / 64), 256, 0, stream>>>(w1 + (size_t)l * D * F, w1T + (size_t)l * F * D, D, F, F);
        transpose_to_f16<<<dim3(D / 64, F / 64), 256, 0, stream>>>(w2 + (size_t)l * F * D, w2T + (size_t)l * D * F, F, D, D);
    }
    transpose_to_f16<<<dim3(VPAD / 64, D / 64), 256, 0, stream>>>(headw, headT, D, V, VPAD);

    // ---- embedding + positional encoding ----
    embed_pe<<<(S * D) / 256, 256, 0, stream>>>(x, emb, h);

    for (int l = 0; l < L; ++l) {
        const _Float16* wqT_ = wqT + (size_t)l * D * D;
        const _Float16* wkT_ = wkT + (size_t)l * D * D;
        const _Float16* wvT_ = wvT + (size_t)l * D * D;
        const _Float16* woT_ = woT + (size_t)l * D * D;
        const _Float16* w1T_ = w1T + (size_t)l * F * D;
        const _Float16* w2T_ = w2T + (size_t)l * D * F;
        const float* bq_ = bq + (size_t)l * D;
        const float* bk_ = bk + (size_t)l * D;
        const float* bv_ = bv + (size_t)l * D;
        const float* bo_ = bo + (size_t)l * D;
        const float* b1_ = b1 + (size_t)l * F;
        const float* b2_ = b2 + (size_t)l * D;

        // QKV projections (f16 async-B weights)
        GEMM(BM_F16T, 0, 1, 0, 0, h, nullptr, wqT_, bq_, nullptr, q, S, D, D, D, D, D);
        GEMM(BM_F16T, 0, 1, 0, 0, h, nullptr, wkT_, bk_, nullptr, k, S, D, D, D, D, D);
        GEMM(BM_F16T, 0, 1, 0, 0, h, nullptr, wvT_, bv_, nullptr, v, S, D, D, D, D, D);

        // Attention, one head at a time (score buffer reused)
        for (int hh = 0; hh < NHEAD; ++hh) {
            const float* qh = q + hh * DHEAD;
            const float* kh = k + hh * DHEAD;
            const float* vh = v + hh * DHEAD;
            // scores = Qh @ Kh^T   [S,S]
            GEMM(BM_F32T, 0, 0, 0, 0, qh, kh, nullptr, nullptr, nullptr, sc,
                 S, S, DHEAD, D, D, S);
            // causal softmax (applies 1/sqrt(DK))
            softmax_causal<<<S, 256, 0, stream>>>(sc, scale);
            // out_h = P @ Vh       [S,DK]
            GEMM(BM_F32N, 0, 0, 0, 0, sc, vh, nullptr, nullptr, nullptr,
                 ao + hh * DHEAD, S, DHEAD, S, S, D, D);
        }

        // O-projection + residual, then LN1
        GEMM(BM_F16T, 0, 1, 1, 0, ao, nullptr, woT_, bo_, h, tmp, S, D, D, D, D, D);
        layernorm_row<<<S, 256, 0, stream>>>(tmp, ln1g + (size_t)l * D,
                                             ln1b + (size_t)l * D, h);

        // FFN: gelu(h@w1+b1) @ w2 + b2 + residual, then LN2
        GEMM(BM_F16T, 1, 1, 0, 0, h, nullptr, w1T_, b1_, nullptr, ff, S, F, D, D, D, F);
        GEMM(BM_F16T, 0, 1, 1, 0, ff, nullptr, w2T_, b2_, h, tmp, S, D, F, F, F, D);
        layernorm_row<<<S, 256, 0, stream>>>(tmp, ln2g + (size_t)l * D,
                                             ln2b + (size_t)l * D, h);
    }

    // Final LN, then LM head (padded f16 weights; epilogue guarded to V)
    layernorm_row<<<S, 256, 0, stream>>>(h, lnfg, lnfb, tmp);
    GEMM(BM_F16T, 0, 1, 0, 1, tmp, nullptr, headT, headb, nullptr, out,
         S, V, D, D, D, V);
}
